// GRD_82300163326471
// MI455X (gfx1250) — compile-verified, matrix operand order checked by reading, not run on hardware
//
#include <hip/hip_runtime.h>
#include <hip/hip_bf16.h>
#include <math.h>

typedef __bf16 bf16_t;
typedef __attribute__((ext_vector_type(16))) __bf16 v16bf;
typedef __attribute__((ext_vector_type(8)))  __bf16 v8bf;
typedef __attribute__((ext_vector_type(8)))  float  v8f;

#define TOPKK 10
#define HID1  512
#define RHID1 150
#define CHC   32
#define NN    150          // nodes
#define BBATCH 32
#define TTIME 100
#define DIN   (NN*CHC)     // 4800
#define G3H1  (3*HID1)     // 1536
#define G3H2  (3*RHID1)    // 450

// ---------------------------------------------------------------------------
// 1) Graph construction: cosine sim -> top-k -> gcn_norm, as dense 150x150.
//    Single block; all phases fenced. Final normalized adjacency lands in A.
// ---------------------------------------------------------------------------
__global__ void build_graph_kernel(const float* __restrict__ emb,
                                   float* __restrict__ A,   // [N*N] scratch -> Mnorm
                                   float* __restrict__ E)   // [N*N] dense edge weights
{
    __shared__ float snorm[NN];
    __shared__ float sdinv[NN];
    int tid = threadIdx.x;
    if (tid < NN) {
        float s = 0.f;
        for (int c = 0; c < CHC; ++c) { float v = emb[tid*CHC + c]; s += v*v; }
        snorm[tid] = fmaxf(sqrtf(s), 1e-8f);
    }
    __syncthreads();
    // cosine similarity, clamp >= 0, zero diagonal; clear E
    for (int e = tid; e < NN*NN; e += blockDim.x) {
        int i = e / NN, j = e % NN;
        float d = 0.f;
        for (int c = 0; c < CHC; ++c) d += emb[i*CHC + c] * emb[j*CHC + c];
        float a = d / (snorm[i] * snorm[j]);
        A[e] = (i == j) ? 0.f : fmaxf(a, 0.f);
        E[e] = 0.f;
    }
    __threadfence_block();
    __syncthreads();
    // per-row top-k; scatter attr into E[dst*N + src]
    if (tid < NN) {
        int chosen[TOPKK];
        for (int k = 0; k < TOPKK; ++k) {
            float mv = -1.f; int mj = 0;
            for (int j = 0; j < NN; ++j) {
                bool used = false;
                for (int p = 0; p < k; ++p) used |= (chosen[p] == j);
                if (used) continue;
                float v = A[tid*NN + j];
                if (v > mv) { mv = v; mj = j; }
            }
            chosen[k] = mj;
            E[mj*NN + tid] = mv;
        }
    }
    __threadfence_block();
    __syncthreads();
    // degree (by dst) and d^{-1/2}
    if (tid < NN) {
        float d = 0.f;
        for (int s = 0; s < NN; ++s) d += E[tid*NN + s];
        sdinv[tid] = (d > 0.f) ? rsqrtf(d) : 0.f;
    }
    __syncthreads();
    // Mnorm[dst][src] = dinv[src] * w * dinv[dst]
    for (int e = tid; e < NN*NN; e += blockDim.x) {
        int dst = e / NN, src = e % NN;
        A[e] = sdinv[src] * E[e] * sdinv[dst];
    }
}

// ---------------------------------------------------------------------------
// 2) ARMA conv + emb add, emitting GRU1 input xt directly in bf16.
//    agg is rank-1: out[n,ch] = gelu(s[n]*w_init[ch] + x[n]*w_root[ch] + b[ch]).
//    One block per (b,t); window row cached in LDS.
// ---------------------------------------------------------------------------
__global__ void arma_xt_kernel(const float* __restrict__ window,
                               const float* __restrict__ emb,
                               const float* __restrict__ w_init,
                               const float* __restrict__ w_root,
                               const float* __restrict__ arma_b,
                               const float* __restrict__ Mnorm,
                               bf16_t* __restrict__ xt)   // [B*T, 4800]
{
    __shared__ float xw[NN];
    int bt  = blockIdx.x;          // b*T + t
    int tid = threadIdx.x;
    if (tid < NN) xw[tid] = window[(size_t)bt*NN + tid];
    __syncthreads();
    if (tid < NN) {
        float x = xw[tid];
        float s = 0.f;
        const float* Mr = Mnorm + tid*NN;
        for (int j = 0; j < NN; ++j) s += Mr[j] * xw[j];
        bf16_t* orow = xt + (size_t)bt*DIN + tid*CHC;
        for (int c = 0; c < CHC; ++c) {
            float pre = s*w_init[c] + x*w_root[c] + arma_b[c];
            float g = 0.5f * pre * (1.f + erff(pre * 0.70710678118654752f));
            orow[c] = (bf16_t)(g + emb[tid*CHC + c]);
        }
    }
}

// ---------------------------------------------------------------------------
// Helpers: fp32 -> bf16 conversion (optionally zero-padded rows)
// ---------------------------------------------------------------------------
__global__ void f2bf_kernel(const float* __restrict__ s, bf16_t* __restrict__ d, int n) {
    int i = blockIdx.x*blockDim.x + threadIdx.x;
    if (i < n) d[i] = (bf16_t)s[i];
}
__global__ void f2bf_pad_kernel(const float* __restrict__ s, bf16_t* __restrict__ d,
                                int rows, int cols, int srows) {
    int i = blockIdx.x*blockDim.x + threadIdx.x;
    if (i < rows*cols) {
        int r = i / cols, c = i % cols;
        d[i] = (r < srows) ? (bf16_t)s[r*cols + c] : (bf16_t)0.f;
    }
}
__global__ void zero_state_kernel(float* h1, bf16_t* h1b, float* h2) {
    int i = blockIdx.x*blockDim.x + threadIdx.x;
    if (i < BBATCH*HID1) { h1[i] = 0.f; h1b[i] = (bf16_t)0.f; }
    if (i < BBATCH*RHID1) h2[i] = 0.f;
}

// ---------------------------------------------------------------------------
// WMMA fragment load: 16-bit A/B operand layout (ISA 7.12.2).
// lane L: row/col = L&15; lanes 0-15 hold K = 0..7 and 16..23,
// lanes 16-31 hold K = 8..15 and 24..31 (two contiguous 16B chunks).
// ---------------------------------------------------------------------------
__device__ __forceinline__ v16bf load_frag(const bf16_t* __restrict__ base, int ld, int lane) {
    int r  = lane & 15;
    int kb = (lane >> 4) << 3;         // 0 or 8
    const bf16_t* p = base + (size_t)r*ld + kb;
    union { v16bf v; v8bf h[2]; } u;
    u.h[0] = *(const v8bf*)(p);        // K = kb .. kb+7
    u.h[1] = *(const v8bf*)(p + 16);   // K = kb+16 .. kb+23
    return u.v;
}

// ---------------------------------------------------------------------------
// 3) Big GEMM: C[M,N] = A[M,K] * B[N,K]^T, bf16 in / f32 out.
//    Block = 8 waves (256 thr), tile 128(M) x 64(N); wave tile 32x32 = 2x2 WMMA.
// ---------------------------------------------------------------------------
__global__ void gemm_bf16_wmma(const bf16_t* __restrict__ A,
                               const bf16_t* __restrict__ B,
                               float* __restrict__ C,
                               int M, int N, int K)
{
    int lane = threadIdx.x & 31;
    int wid  = threadIdx.x >> 5;
    int m0 = blockIdx.y*128 + (wid >> 1)*32;
    int n0 = blockIdx.x*64  + (wid & 1)*32;
    v8f acc00 = {}, acc01 = {}, acc10 = {}, acc11 = {};
    const bf16_t* Ar0 = A + (size_t)m0*K;
    const bf16_t* Ar1 = A + (size_t)(m0+16)*K;
    const bf16_t* Br0 = B + (size_t)n0*K;
    const bf16_t* Br1 = B + (size_t)(n0+16)*K;
    for (int k = 0; k < K; k += 32) {
        __builtin_prefetch(Ar0 + k + 256, 0, 1);   // global_prefetch_b8
        __builtin_prefetch(Br0 + k + 256, 0, 1);
        v16bf a0 = load_frag(Ar0 + k, K, lane);
        v16bf a1 = load_frag(Ar1 + k, K, lane);
        v16bf b0 = load_frag(Br0 + k, K, lane);
        v16bf b1 = load_frag(Br1 + k, K, lane);
        acc00 = __builtin_amdgcn_wmma_f32_16x16x32_bf16(false, a0, false, b0, (short)0, acc00, false, false);
        acc01 = __builtin_amdgcn_wmma_f32_16x16x32_bf16(false, a0, false, b1, (short)0, acc01, false, false);
        acc10 = __builtin_amdgcn_wmma_f32_16x16x32_bf16(false, a1, false, b0, (short)0, acc10, false, false);
        acc11 = __builtin_amdgcn_wmma_f32_16x16x32_bf16(false, a1, false, b1, (short)0, acc11, false, false);
    }
    // C layout: VGPR i holds M = i (lanes 0-15) / i+8 (lanes 16-31); N = lane&15
    int col = lane & 15, half = lane >> 4;
    for (int i = 0; i < 8; ++i) {
        int mA = m0 + half*8 + i;
        int mB = m0 + 16 + half*8 + i;
        C[(size_t)mA*N + n0 + col]      = acc00[i];
        C[(size_t)mA*N + n0 + 16 + col] = acc01[i];
        C[(size_t)mB*N + n0 + col]      = acc10[i];
        C[(size_t)mB*N + n0 + 16 + col] = acc11[i];
    }
}

// Recurrent GEMM: C[32,N] = A[32,K] * B[N,K]^T. One wave = 32x32 tile.
__global__ void gemm_bf16_wmma_m32(const bf16_t* __restrict__ A,
                                   const bf16_t* __restrict__ B,
                                   float* __restrict__ C,
                                   int N, int K)
{
    int lane = threadIdx.x & 31;
    int wid  = threadIdx.x >> 5;
    int n0 = blockIdx.x*256 + wid*32;
    if (n0 >= N) return;
    v8f acc00 = {}, acc01 = {}, acc10 = {}, acc11 = {};
    const bf16_t* Ar1 = A + (size_t)16*K;
    const bf16_t* Br0 = B + (size_t)n0*K;
    const bf16_t* Br1 = B + (size_t)(n0+16)*K;
    for (int k = 0; k < K; k += 32) {
        v16bf a0 = load_frag(A   + k, K, lane);
        v16bf a1 = load_frag(Ar1 + k, K, lane);
        v16bf b0 = load_frag(Br0 + k, K, lane);
        v16bf b1 = load_frag(Br1 + k, K, lane);
        acc00 = __builtin_amdgcn_wmma_f32_16x16x32_bf16(false, a0, false, b0, (short)0, acc00, false, false);
        acc01 = __builtin_amdgcn_wmma_f32_16x16x32_bf16(false, a0, false, b1, (short)0, acc01, false, false);
        acc10 = __builtin_amdgcn_wmma_f32_16x16x32_bf16(false, a1, false, b0, (short)0, acc10, false, false);
        acc11 = __builtin_amdgcn_wmma_f32_16x16x32_bf16(false, a1, false, b1, (short)0, acc11, false, false);
    }
    int col = lane & 15, half = lane >> 4;
    for (int i = 0; i < 8; ++i) {
        int mA = half*8 + i;
        int mB = 16 + half*8 + i;
        C[(size_t)mA*N + n0 + col]      = acc00[i];
        C[(size_t)mA*N + n0 + 16 + col] = acc01[i];
        C[(size_t)mB*N + n0 + col]      = acc10[i];
        C[(size_t)mB*N + n0 + 16 + col] = acc11[i];
    }
}

// ---------------------------------------------------------------------------
// 4) GRU gate update (torch gate order r,z,n). Biases added here.
// ---------------------------------------------------------------------------
__global__ void gru_gate_kernel(const float* __restrict__ xi, int ld_xi,  // rows = b*T+t
                                const float* __restrict__ gh,             // [B, 3H]
                                const float* __restrict__ bi,
                                const float* __restrict__ bh,
                                float* __restrict__ h,                    // [B,H] in/out
                                bf16_t* __restrict__ hbf,                 // optional bf16 mirror
                                float* __restrict__ dec,                  // optional [B*T,H] seq out
                                int H, int t)
{
    int i = blockIdx.x*blockDim.x + threadIdx.x;
    if (i >= BBATCH*H) return;
    int b = i / H, j = i % H;
    const float* xr = xi + (size_t)(b*TTIME + t)*ld_xi;
    const float* gr = gh + (size_t)b*3*H;
    float r  = xr[j]       + bi[j]       + gr[j]       + bh[j];
    float z  = xr[H+j]     + bi[H+j]     + gr[H+j]     + bh[H+j];
    float nx = xr[2*H+j]   + bi[2*H+j];
    float gn = gr[2*H+j]   + bh[2*H+j];
    r = 1.f / (1.f + expf(-r));
    z = 1.f / (1.f + expf(-z));
    float n = tanhf(nx + r*gn);
    float hn = (1.f - z)*n + z*h[i];
    h[i] = hn;
    if (hbf) hbf[i] = (bf16_t)hn;
    if (dec) dec[(size_t)(b*TTIME + t)*H + j] = hn;
}

// repeat_interleave(T,dim=1).view(B,T,HID): hrep[b,t,k] = hend[b,(t*HID+k)/T]
__global__ void hrep_kernel(const float* __restrict__ hend, bf16_t* __restrict__ hrep) {
    int i = blockIdx.x*blockDim.x + threadIdx.x;
    if (i >= BBATCH*TTIME*HID1) return;
    int k = i % HID1;
    int bt = i / HID1;
    int t = bt % TTIME, b = bt / TTIME;
    hrep[i] = (bf16_t)hend[b*HID1 + (t*HID1 + k)/TTIME];
}

// GRU2 recurrent projection (tiny): gh2[b,m] = h2[b,:] . Wh2[m,:]
__global__ void gh2_kernel(const float* __restrict__ h2, const float* __restrict__ Wh2,
                           float* __restrict__ gh2) {
    int i = blockIdx.x*blockDim.x + threadIdx.x;
    if (i >= BBATCH*G3H2) return;
    int b = i / G3H2, m = i % G3H2;
    const float* hv = h2 + b*RHID1;
    const float* w  = Wh2 + m*RHID1;
    float s = 0.f;
    for (int k = 0; k < RHID1; ++k) s += hv[k]*w[k];
    gh2[i] = s;
}

// 5) Final FC: out[bt,n] = b_fc[n] + dec[bt,:] . W_fc[n,:]
__global__ void fc_kernel(const float* __restrict__ dec, const float* __restrict__ Wfc,
                          const float* __restrict__ bfc, float* __restrict__ out) {
    int i = blockIdx.x*blockDim.x + threadIdx.x;
    if (i >= BBATCH*TTIME*NN) return;
    int n = i % NN, bt = i / NN;
    const float* dv = dec + (size_t)bt*RHID1;
    const float* w  = Wfc + n*RHID1;
    float s = bfc[n];
    for (int k = 0; k < RHID1; ++k) s += dv[k]*w[k];
    out[i] = s;
}

// ---------------------------------------------------------------------------
extern "C" void kernel_launch(void* const* d_in, const int* in_sizes, int n_in,
                              void* d_out, int out_size, void* d_ws, size_t ws_size,
                              hipStream_t stream) {
    const float* window = (const float*)d_in[0];
    const float* emb    = (const float*)d_in[1];
    const float* w_init = (const float*)d_in[2];
    const float* w_root = (const float*)d_in[3];
    const float* arma_b = (const float*)d_in[4];
    const float* Wi1    = (const float*)d_in[5];
    const float* Wh1    = (const float*)d_in[6];
    const float* bi1    = (const float*)d_in[7];
    const float* bh1    = (const float*)d_in[8];
    const float* Wi2    = (const float*)d_in[9];
    const float* Wh2    = (const float*)d_in[10];
    const float* bi2    = (const float*)d_in[11];
    const float* bh2    = (const float*)d_in[12];
    const float* W_fc   = (const float*)d_in[13];
    const float* b_fc   = (const float*)d_in[14];
    float* out = (float*)d_out;

    // --- workspace layout (256B aligned) ---
    char* base = (char*)d_ws;
    size_t off = 0;
    auto take = [&](size_t bytes) { char* p = base + off; off += (bytes + 255) & ~(size_t)255; return p; };
    const size_t MBT = (size_t)BBATCH * TTIME;         // 3200
    float*  bufA  = (float*) take((size_t)NN*NN*4);
    float*  bufE  = (float*) take((size_t)NN*NN*4);
    bf16_t* xtbf  = (bf16_t*)take(MBT*DIN*2);          // [3200,4800]
    bf16_t* Wi1bf = (bf16_t*)take((size_t)G3H1*DIN*2); // [1536,4800]
    float*  xi1   = (float*) take(MBT*G3H1*4);         // [3200,1536]
    bf16_t* Wh1bf = (bf16_t*)take((size_t)G3H1*HID1*2);
    float*  h1    = (float*) take((size_t)BBATCH*HID1*4);
    bf16_t* h1bf  = (bf16_t*)take((size_t)BBATCH*HID1*2);
    float*  gh1   = (float*) take((size_t)BBATCH*G3H1*4);
    bf16_t* hrep  = (bf16_t*)take(MBT*HID1*2);         // [3200,512]
    bf16_t* Wi2p  = (bf16_t*)take((size_t)512*512*2);  // Wi2 zero-padded 450->512 rows
    float*  xi2   = (float*) take(MBT*512*4);          // padded N
    float*  h2    = (float*) take((size_t)BBATCH*RHID1*4);
    float*  gh2   = (float*) take((size_t)BBATCH*G3H2*4);
    float*  dec   = (float*) take(MBT*RHID1*4);
    (void)ws_size; (void)in_sizes; (void)n_in; (void)out_size;

    // init recurrent state
    zero_state_kernel<<<(BBATCH*HID1 + 255)/256, 256, 0, stream>>>(h1, h1bf, h2);
    // graph + ARMA feature build (emits xt in bf16)
    build_graph_kernel<<<1, 256, 0, stream>>>(emb, bufA, bufE);
    arma_xt_kernel<<<(int)MBT, 160, 0, stream>>>(window, emb, w_init, w_root, arma_b, bufA, xtbf);
    // weight conversions
    f2bf_kernel<<<(G3H1*DIN + 255)/256, 256, 0, stream>>>(Wi1, Wi1bf, G3H1*DIN);
    f2bf_kernel<<<(G3H1*HID1 + 255)/256, 256, 0, stream>>>(Wh1, Wh1bf, G3H1*HID1);
    f2bf_pad_kernel<<<(512*512 + 255)/256, 256, 0, stream>>>(Wi2, Wi2p, 512, 512, G3H2);
    // big input projection: xi1 = xt @ Wi1^T  (47 GFLOP, WMMA bf16)
    gemm_bf16_wmma<<<dim3(G3H1/64, (int)MBT/128), 256, 0, stream>>>(xtbf, Wi1bf, xi1, (int)MBT, G3H1, DIN);
    // GRU1 over time (sequential)
    for (int t = 0; t < TTIME; ++t) {
        gemm_bf16_wmma_m32<<<G3H1/256, 256, 0, stream>>>(h1bf, Wh1bf, gh1, G3H1, HID1);
        gru_gate_kernel<<<(BBATCH*HID1 + 255)/256, 256, 0, stream>>>(
            xi1, G3H1, gh1, bi1, bh1, h1, h1bf, nullptr, HID1, t);
    }
    // decoder input: faithful repeat_interleave indexing, then xi2 = hrep @ Wi2^T
    hrep_kernel<<<((int)(MBT*HID1) + 255)/256, 256, 0, stream>>>(h1, hrep);
    gemm_bf16_wmma<<<dim3(512/64, (int)MBT/128), 256, 0, stream>>>(hrep, Wi2p, xi2, (int)MBT, 512, HID1);
    // GRU2 over time, writing full sequence into dec
    for (int t = 0; t < TTIME; ++t) {
        gh2_kernel<<<(BBATCH*G3H2 + 255)/256, 256, 0, stream>>>(h2, Wh2, gh2);
        gru_gate_kernel<<<(BBATCH*RHID1 + 255)/256, 256, 0, stream>>>(
            xi2, 512, gh2, bi2, bh2, h2, nullptr, dec, RHID1, t);
    }
    // output projection
    fc_kernel<<<((int)(MBT*NN) + 255)/256, 256, 0, stream>>>(dec, W_fc, b_fc, out);
}
